// GRU_5995774345610
// MI455X (gfx1250) — compile-verified, matrix operand order checked by reading, not run on hardware
//
#include <hip/hip_runtime.h>
#include <hip/hip_bf16.h>
#include <cstdint>
#include <cstddef>

// ---------------------------------------------------------------------------
// Problem constants (from reference): B=64, T=1024, IN=EMB=HID=OUT=512, L=4
// ---------------------------------------------------------------------------
#define BB   64
#define TT   1024
#define EMB  512
#define HID  512
#define LAYERS 4
#define G3H  (3 * HID)          // 1536
#define MTOK (BB * TT)          // 65536 tokens

typedef __bf16 bf16_t;
typedef __attribute__((ext_vector_type(16))) __bf16 v16bf;
typedef __attribute__((ext_vector_type(8)))  float  v8f;

union FragU { v16bf v; uint4 u[2]; };
union AccU  { v8f v; float f[8]; };

// ---------------------------------------------------------------------------
// CDNA5 async memory->LDS copy (ASYNCcnt-tracked, no VGPR data movement).
// Falls back to a synchronous VGPR round-trip if the builtin is unavailable.
// ---------------------------------------------------------------------------
#if __has_builtin(__builtin_amdgcn_global_load_async_to_lds_b128)
#define HAVE_ASYNC_LDS 1
#else
#define HAVE_ASYNC_LDS 0
#endif

typedef int v4i_vs __attribute__((vector_size(16)));   // matches builtin's V4i

__device__ __forceinline__ void async_copy_b128(const uint16_t* gsrc,
                                                uint16_t* ldst) {
#if HAVE_ASYNC_LDS
  __builtin_amdgcn_global_load_async_to_lds_b128(
      (__attribute__((address_space(1))) v4i_vs*)(uintptr_t)gsrc,
      (__attribute__((address_space(3))) v4i_vs*)(uint32_t)(uintptr_t)ldst,
      /*offset=*/0, /*cpol=*/0);
#else
  *(uint4*)ldst = *(const uint4*)gsrc;
#endif
}

__device__ __forceinline__ void wait_async_all() {
#if HAVE_ASYNC_LDS
#if __has_builtin(__builtin_amdgcn_s_wait_asynccnt)
  __builtin_amdgcn_s_wait_asynccnt(0);
#else
  asm volatile("s_wait_asynccnt 0" ::: "memory");
#endif
#endif
}

__device__ __forceinline__ uint16_t f32_to_bf16(float f) {
  uint32_t x = __float_as_uint(f);
  uint32_t r = x + 0x7FFFu + ((x >> 16) & 1u);   // round to nearest even
  return (uint16_t)(r >> 16);
}
__device__ __forceinline__ float bf16_to_f32(uint16_t u) {
  return __uint_as_float(((uint32_t)u) << 16);
}
__device__ __forceinline__ float sigmoidf_fast(float x) {
  return 1.0f / (1.0f + __expf(-x));
}
__device__ __forceinline__ float tanhf_fast(float x) {
  float xx = x > 15.f ? 15.f : (x < -15.f ? -15.f : x);
  float e = __expf(-2.0f * xx);
  return (1.0f - e) / (1.0f + e);
}

// Load a 16x32 bf16 WMMA fragment (A-layout; B-layout identical with
// row<->col roles swapped): lane L: row = L&15, half = L>>4; two contiguous
// 16B chunks at K = k0+half*8 and k0+16+half*8.
__device__ __forceinline__ v16bf load_frag(const uint16_t* __restrict__ base,
                                           int ld, int row0, int k0, int lane) {
  int hf = lane >> 4;
  int r  = lane & 15;
  const uint16_t* p = base + (size_t)(row0 + r) * ld + k0 + hf * 8;
  FragU f;
  f.u[0] = *(const uint4*)(p);
  f.u[1] = *(const uint4*)(p + 16);
  return f.v;
}

__device__ __forceinline__ v8f wmma_bf16(v16bf a, v16bf b, v8f c) {
  return __builtin_amdgcn_wmma_f32_16x16x32_bf16(
      /*neg_a=*/false, a, /*neg_b=*/false, b,
      /*c_mod=*/(short)0, c, /*reuse_a=*/false, /*reuse_b=*/false);
}

// ---------------------------------------------------------------------------
// Utility kernels
// ---------------------------------------------------------------------------
__global__ void k_f32_to_bf16(const float* __restrict__ src,
                              uint16_t* __restrict__ dst, size_t n) {
  size_t i = (size_t)blockIdx.x * blockDim.x + threadIdx.x;
  size_t stride = (size_t)gridDim.x * blockDim.x;
  for (; i < n; i += stride) dst[i] = f32_to_bf16(src[i]);
}

__global__ void k_zero_u32(uint32_t* __restrict__ p, size_t n) {
  size_t i = (size_t)blockIdx.x * blockDim.x + threadIdx.x;
  size_t stride = (size_t)gridDim.x * blockDim.x;
  for (; i < n; i += stride) p[i] = 0u;
}

// ---------------------------------------------------------------------------
// GEMM: Y[M,N] = act(X[M,K](bf16) @ W[N,K](bf16)^T + bias)
// WG = 128 threads (4 waves), tile 128(M) x 64(N).
// The WG's 64 weight rows x K are async-staged into LDS once (64KB bf16) and
// shared by all waves; each wave register-blocks 32 rows x 64 cols
// (8 f32 accumulators): each B fragment (ds_load_b128) feeds 2 WMMAs and each
// A fragment feeds 4 WMMAs.
// ---------------------------------------------------------------------------
template <bool RELU, bool OUT_BF16>
__global__ __launch_bounds__(128)
void k_gemm_bias_act(const uint16_t* __restrict__ X,
                     const uint16_t* __restrict__ W,
                     const float* __restrict__ bias,
                     uint16_t* __restrict__ Yb,
                     float* __restrict__ Yf,
                     int M, int N, int K) {
  extern __shared__ uint16_t wlds[];   // [64][K] bf16
  const int wave = threadIdx.x >> 5;
  const int lane = threadIdx.x & 31;
  const int ntiles = N >> 6;
  const int m0 = (blockIdx.x / ntiles) * 128 + wave * 32;
  const int n0 = (blockIdx.x % ntiles) * 64;

  // Async-stage W rows [n0, n0+64) x K into LDS.
  for (int idx = threadIdx.x * 8; idx < 64 * K; idx += blockDim.x * 8) {
    int r = idx / K, k = idx - r * K;
    async_copy_b128(W + (size_t)(n0 + r) * K + k, wlds + idx);
  }
  wait_async_all();
  __syncthreads();

  v8f acc[2][4] = {{v8f{}, v8f{}, v8f{}, v8f{}}, {v8f{}, v8f{}, v8f{}, v8f{}}};

  for (int k0 = 0; k0 < K; k0 += 32) {
    if (k0 + 32 < K)  // pull next A chunks toward the WGP (global_prefetch_b8)
      __builtin_prefetch(X + (size_t)m0 * K + k0 + 32, 0, 1);
    v16bf a0 = load_frag(X, K, m0, k0, lane);
    v16bf a1 = load_frag(X, K, m0 + 16, k0, lane);
#pragma unroll
    for (int nt = 0; nt < 4; ++nt) {
      v16bf b = load_frag(wlds, K, nt * 16, k0, lane);  // ds_load_b128
      acc[0][nt] = wmma_bf16(a0, b, acc[0][nt]);
      acc[1][nt] = wmma_bf16(a1, b, acc[1][nt]);
    }
  }

  const int hf = lane >> 4, col = lane & 15;
#pragma unroll
  for (int mt = 0; mt < 2; ++mt) {
#pragma unroll
    for (int nt = 0; nt < 4; ++nt) {
      int n = n0 + nt * 16 + col;
      float bv = bias[n];
      AccU A; A.v = acc[mt][nt];
#pragma unroll
      for (int r = 0; r < 8; ++r) {
        int row = m0 + mt * 16 + r + hf * 8;
        float v = A.f[r] + bv;
        if (RELU) v = v > 0.f ? v : 0.f;
        if (OUT_BF16) Yb[(size_t)row * N + n] = f32_to_bf16(v);
        else          Yf[(size_t)row * N + n] = v;
      }
    }
  }
}

// ---------------------------------------------------------------------------
// GRU recurrent scan for one layer.
// Grid: 32 WGs = 4 batch-tiles(m) x 8 column-chunks(s). Block: 128 (4 waves).
// W_hh slab (3 gates x 64 cols x 512 K bf16 = 192KB) async-staged once into
// the 320KB WGP LDS, resident for all T steps. Per step: async-stage the
// h[16,512] tile AND the gi[16,192] gate tile to LDS (overlapped, one
// s_wait_asynccnt), 48 WMMAs/wave from pure LDS operands, gate math, write h'
// to the flip global h buffer, release/acquire counter sync across the 8
// column-chunk WGs of this batch tile.
// ---------------------------------------------------------------------------
__global__ __launch_bounds__(128)
void k_gru_scan(const uint16_t* __restrict__ gi,    // [B,T,3H] bf16 (has b_ih)
                const uint16_t* __restrict__ whh,   // [3H,H]   bf16
                const float*    __restrict__ bhh,   // [3H]     f32
                uint16_t* __restrict__ hstate,      // [2][B][H] bf16 (zeroed)
                uint16_t* __restrict__ yout,        // [B,T,H]  bf16
                uint32_t* __restrict__ cnt,         // [4] per-m counters (0)
                int T) {
  constexpr int H = HID;
  constexpr int NS = 8;                 // column-chunk WGs per batch tile
  extern __shared__ uint16_t lds[];
  uint16_t* whh_lds = lds;                        // 3*64*H bf16 (192KB)
  uint16_t* hb      = lds + 3 * 64 * H;           // 16*H bf16   (16KB)
  uint16_t* gib     = lds + 3 * 64 * H + 16 * H;  // 16*192 bf16 (6KB)

  const int m = blockIdx.x / NS;        // batch tile (16 rows)
  const int s = blockIdx.x % NS;        // 64-col hidden chunk
  const int wave = threadIdx.x >> 5;
  const int lane = threadIdx.x & 31;
  const int b0 = m * 16;
  const int jloc = wave * 16;           // col-tile inside chunk (0..48)
  const int j0 = s * 64 + jloc;         // global hidden col base
  const int hf = lane >> 4, col = lane & 15;

  // Async-stage W_hh rows {g*H + s*64 .. +64}, g=0..2, into LDS.
  for (int idx = threadIdx.x * 8; idx < 3 * 64 * H; idx += blockDim.x * 8) {
    int row = idx >> 9;                 // /H (H=512)
    int k   = idx & (H - 1);
    int g = row >> 6, c = row & 63;
    async_copy_b128(whh + (size_t)(g * H + s * 64 + c) * H + k, whh_lds + idx);
  }
  // Per-lane hidden-side biases (fixed column across all steps).
  const float br = bhh[0 * H + j0 + col];
  const float bz = bhh[1 * H + j0 + col];
  const float bn = bhh[2 * H + j0 + col];
  wait_async_all();
  __syncthreads();

  for (int t = 0; t < T; ++t) {
    const int p = t & 1;
    const uint16_t* hprev = hstate + (size_t)p * BB * H;
    uint16_t* hnext = hstate + (size_t)(p ^ 1) * BB * H;

    // Async-stage h tile [16][H] into LDS (1024 b128 ops, 8/thread).
    for (int idx = threadIdx.x * 8; idx < 16 * H; idx += blockDim.x * 8) {
      int r = idx >> 9;
      int k = idx & (H - 1);
      async_copy_b128(hprev + (size_t)(b0 + r) * H + k, hb + idx);
    }
    // Async-stage gi tile [16 rows][3 gates][64 cols] (384 b128, 3/thread).
    for (int idx = threadIdx.x; idx < 16 * 3 * 8; idx += blockDim.x) {
      int row = idx / 24;
      int rem = idx - row * 24;
      int g = rem >> 3;
      int c8 = (rem & 7) * 8;
      const uint16_t* src =
          gi + ((size_t)(b0 + row) * T + t) * (size_t)G3H + g * H + s * 64 + c8;
      async_copy_b128(src, gib + row * 192 + g * 64 + c8);
    }
    wait_async_all();
    __syncthreads();

    // gh = h @ Whh^T for the 3 gate tiles, all operands from LDS.
    v8f ar = {}, az = {}, an = {};
    for (int k0 = 0; k0 < H; k0 += 32) {
      v16bf a = load_frag(hb, H, 0, k0, lane);
      ar = wmma_bf16(a, load_frag(whh_lds, H, 0 * 64 + jloc, k0, lane), ar);
      az = wmma_bf16(a, load_frag(whh_lds, H, 1 * 64 + jloc, k0, lane), az);
      an = wmma_bf16(a, load_frag(whh_lds, H, 2 * 64 + jloc, k0, lane), an);
    }

    AccU R, Z, Nn; R.v = ar; Z.v = az; Nn.v = an;
#pragma unroll
    for (int r = 0; r < 8; ++r) {
      int row = r + hf * 8;                       // 0..15 in tile
      int lc  = jloc + col;                       // 0..63 in chunk
      float ir  = bf16_to_f32(gib[row * 192 + 0 * 64 + lc]);
      float iz  = bf16_to_f32(gib[row * 192 + 1 * 64 + lc]);
      float inn = bf16_to_f32(gib[row * 192 + 2 * 64 + lc]);
      float rg = sigmoidf_fast(ir + R.f[r] + br);
      float zg = sigmoidf_fast(iz + Z.f[r] + bz);
      float ng = tanhf_fast(inn + rg * (Nn.f[r] + bn));
      float hold = bf16_to_f32(hb[row * H + j0 + col]);
      float hnew = (1.0f - zg) * ng + zg * hold;
      uint16_t h16 = f32_to_bf16(hnew);
      hnext[(size_t)(b0 + row) * H + j0 + col] = h16;
      yout[((size_t)(b0 + row) * T + t) * H + j0 + col] = h16;
    }

    __threadfence();          // release h' writes to device scope
    __syncthreads();          // all lanes done with hb/gib for this step
    if (threadIdx.x == 0)
      __hip_atomic_fetch_add(&cnt[m], 1u, __ATOMIC_RELEASE,
                             __HIP_MEMORY_SCOPE_AGENT);
    const uint32_t target = (uint32_t)(NS * (t + 1));
    while (__hip_atomic_load(&cnt[m], __ATOMIC_ACQUIRE,
                             __HIP_MEMORY_SCOPE_AGENT) < target) {
      __builtin_amdgcn_s_sleep(1);
    }
  }
}

// ---------------------------------------------------------------------------
// Host-side orchestration
// ---------------------------------------------------------------------------
static inline size_t align256(size_t x) { return (x + 255) & ~(size_t)255; }

extern "C" void kernel_launch(void* const* d_in, const int* in_sizes, int n_in,
                              void* d_out, int out_size, void* d_ws, size_t ws_size,
                              hipStream_t stream) {
  (void)in_sizes; (void)n_in; (void)out_size; (void)ws_size;
  const float* x    = (const float*)d_in[0];   // [B,T,IN]
  const float* w_in = (const float*)d_in[1];   // [EMB,IN]
  const float* b_in = (const float*)d_in[2];   // [EMB]
  const float* w_ih = (const float*)d_in[3];   // [L,3H,EMB]
  const float* w_hh = (const float*)d_in[4];   // [L,3H,HID]
  const float* b_ih = (const float*)d_in[5];   // [L,3H]
  const float* b_hh = (const float*)d_in[6];   // [L,3H]
  const float* w_o1 = (const float*)d_in[7];   // [EMB,HID]
  const float* b_o1 = (const float*)d_in[8];   // [EMB]
  const float* w_o2 = (const float*)d_in[9];   // [OUT,EMB]
  const float* b_o2 = (const float*)d_in[10];  // [OUT]
  float* out = (float*)d_out;                  // [B,T,OUT]

  // -------- workspace layout (bf16 stored as uint16_t) --------
  char* ws = (char*)d_ws;
  size_t off = 0;
  auto take = [&](size_t elems2b) {
    void* p = ws + off;
    off = align256(off + elems2b * 2);
    return p;
  };
  uint16_t* xb    = (uint16_t*)take((size_t)MTOK * EMB);       // x bf16
  uint16_t* wbin  = (uint16_t*)take((size_t)EMB * EMB);
  uint16_t* wbih  = (uint16_t*)take((size_t)LAYERS * G3H * EMB);
  uint16_t* wbhh  = (uint16_t*)take((size_t)LAYERS * G3H * HID);
  uint16_t* wbo1  = (uint16_t*)take((size_t)EMB * HID);
  uint16_t* wbo2  = (uint16_t*)take((size_t)EMB * EMB);
  uint16_t* hA    = (uint16_t*)take((size_t)MTOK * EMB);       // ping
  uint16_t* hB    = (uint16_t*)take((size_t)MTOK * EMB);       // pong
  uint16_t* gi    = (uint16_t*)take((size_t)MTOK * G3H);       // gate pre-acts
  uint16_t* hstate= (uint16_t*)take((size_t)2 * BB * HID);     // dbl-buffered h
  uint32_t* cnt   = (uint32_t*)(ws + off);
  off = align256(off + LAYERS * 4 * sizeof(uint32_t));

  // -------- 1) convert inputs/weights to bf16 --------
  auto cvt = [&](const float* s, uint16_t* d, size_t n) {
    int blocks = (int)((n + 8 * 256 - 1) / (8 * 256));
    if (blocks > 4096) blocks = 4096;
    k_f32_to_bf16<<<blocks, 256, 0, stream>>>(s, d, n);
  };
  cvt(x,    xb,   (size_t)MTOK * EMB);
  cvt(w_in, wbin, (size_t)EMB * EMB);
  cvt(w_ih, wbih, (size_t)LAYERS * G3H * EMB);
  cvt(w_hh, wbhh, (size_t)LAYERS * G3H * HID);
  cvt(w_o1, wbo1, (size_t)EMB * HID);
  cvt(w_o2, wbo2, (size_t)EMB * EMB);
  // zero sync counters once per call
  k_zero_u32<<<1, 64, 0, stream>>>(cnt, LAYERS * 4);

  const size_t GEMM_LDS  = (size_t)64 * 512 * 2;                       // 64KB
  const size_t SCAN_LDS  = (size_t)(3 * 64 * HID + 16 * HID + 16 * 192) * 2;

  // -------- 2) input Linear + ReLU -> hA --------
  k_gemm_bias_act<true, true><<<(MTOK / 128) * (EMB / 64), 128, GEMM_LDS,
                                stream>>>(
      xb, wbin, b_in, hA, nullptr, MTOK, EMB, EMB);

  // -------- 3) stacked GRU --------
  uint16_t* cur = hA;
  uint16_t* nxt = hB;
  for (int l = 0; l < LAYERS; ++l) {
    // gate pre-activations gi = cur @ W_ih[l]^T + b_ih[l] (full-GPU GEMM)
    k_gemm_bias_act<false, true><<<(MTOK / 128) * (G3H / 64), 128, GEMM_LDS,
                                   stream>>>(
        cur, wbih + (size_t)l * G3H * EMB, b_ih + (size_t)l * G3H,
        gi, nullptr, MTOK, G3H, EMB);
    // h0 = 0 for this layer
    k_zero_u32<<<64, 256, 0, stream>>>((uint32_t*)hstate,
                                       (size_t)2 * BB * HID / 2);
    // latency-critical recurrent scan: 32 WGs, W_hh LDS-stationary
    k_gru_scan<<<32, 128, SCAN_LDS, stream>>>(
        gi, wbhh + (size_t)l * G3H * HID, b_hh + (size_t)l * G3H,
        hstate, nxt, cnt + l * 4, TT);
    uint16_t* tmp = cur; cur = nxt; nxt = tmp;
  }

  // -------- 4) output MLP: Linear+ReLU -> Linear (f32 out) --------
  uint16_t* t1 = gi;  // reuse gi buffer for the MLP intermediate
  k_gemm_bias_act<true, true><<<(MTOK / 128) * (EMB / 64), 128, GEMM_LDS,
                                stream>>>(
      cur, wbo1, b_o1, t1, nullptr, MTOK, EMB, HID);
  k_gemm_bias_act<false, false><<<(MTOK / 128) * (EMB / 64), 128, GEMM_LDS,
                                  stream>>>(
      t1, wbo2, b_o2, nullptr, out, MTOK, EMB, EMB);
}
